// RotaryMultiheadAttention_27934467293739
// MI455X (gfx1250) — compile-verified, matrix operand order checked by reading, not run on hardware
//
#include <hip/hip_runtime.h>
#include <hip/hip_bf16.h>

// ---------------------------------------------------------------------------
// RotaryMultiheadAttention forward for MI455X (gfx1250, wave32, WMMA bf16).
// B=2, S=2048, D=1024, H=16, HD=64.  All GEMMs via v_wmma_f32_16x16x32_bf16.
// Round 4: all staged operands stored in fragment-native swizzled layout so
// every WMMA operand is ONE contiguous 32-byte load (no v16bf packing movs).
//
// Swizzle (per ISA 7.12.2, 16-bit A/B operands): a 16(row)x32(col) bf16 tile
// occupies 512 elements; lane = (row&15) + 16*((col>>3)&1),
// elem = (col&7) + 8*((col>>4)&1); fragment = 16 contiguous elems per lane.
// ---------------------------------------------------------------------------

typedef __attribute__((ext_vector_type(16))) __bf16 v16bf;
typedef __attribute__((ext_vector_type(8)))  float  v8f;

static __device__ __forceinline__ v8f wmma_bf16(v16bf a, v16bf b, v8f c) {
    return __builtin_amdgcn_wmma_f32_16x16x32_bf16(
        /*neg_a=*/false, a, /*neg_b=*/false, b,
        /*c_mod=*/(short)0, c, /*reuse_a=*/false, /*reuse_b=*/false);
}

// element index inside a swizzled matrix with Kdim columns (Kdim % 32 == 0)
static __device__ __forceinline__ size_t swz_idx(int row, int col, int Kdim) {
    const int tile = (row >> 4) * (Kdim >> 5) + (col >> 5);
    const int lane = (row & 15) + (((col >> 3) & 1) << 4);
    const int e    = (col & 7) + (((col >> 4) & 1) << 3);
    return (size_t)tile * 512 + lane * 16 + e;
}

static __device__ __forceinline__ float gelu_exact(float x) {
    return 0.5f * x * (1.0f + erff(x * 0.70710678118654752f));
}

// ---------------------------------------------------------------------------
// fp32 row-major (M x 1024) -> bf16 swizzled.  One 8-col chunk per thread:
// chunk stays contiguous in the swizzled layout -> single 16B store.
// ---------------------------------------------------------------------------
__global__ void cvt_swz(const float* __restrict__ src, __bf16* __restrict__ dst,
                        int nchunks) {
    int c = blockIdx.x * blockDim.x + threadIdx.x;
    if (c >= nchunks) return;
    const int m  = c >> 7;            // Kdim = 1024 -> 128 chunks per row
    const int kc = (c & 127) << 3;
    const float* s = src + (size_t)m * 1024 + kc;
    float4 x0 = *(const float4*)s;
    float4 x1 = *(const float4*)(s + 4);
    union { __bf16 h[8]; uint4 u; } pk;
    pk.h[0] = (__bf16)x0.x; pk.h[1] = (__bf16)x0.y;
    pk.h[2] = (__bf16)x0.z; pk.h[3] = (__bf16)x0.w;
    pk.h[4] = (__bf16)x1.x; pk.h[5] = (__bf16)x1.y;
    pk.h[6] = (__bf16)x1.z; pk.h[7] = (__bf16)x1.w;
    const int tile = (m >> 4) * 32 + (kc >> 5);
    const int lane = (m & 15) + (((kc >> 3) & 1) << 4);
    const int e0   = ((kc >> 4) & 1) << 3;
    *(uint4*)(dst + (size_t)tile * 512 + lane * 16 + e0) = pk.u;
}

// ---------------------------------------------------------------------------
// GEMM: out = epilogue( A(4096x1024) @ W(1024x1024 row=out)^T + b ), both
// operands bf16-swizzled.  One wave = 16x64 tile; K=1024 as 32 WMMA steps,
// ping-pong buffered; every fragment is a single 32B load.
// mode 0: gelu(+rope) -> Qb/Kb per-head [t][64]  swizzled
// mode 1: gelu        -> Vt  per-head [64][t]    swizzled (for P@V)
// mode 2: gelu        -> out fp32 row-major (identity onto (B,S,D) flat)
// ---------------------------------------------------------------------------
__global__ __launch_bounds__(256) void proj_gemm(
    const __bf16* __restrict__ A, const __bf16* __restrict__ W,
    const float* __restrict__ bias,
    __bf16* __restrict__ out_bf, float* __restrict__ out_f,
    int mode, int do_rope) {
    const int tid = threadIdx.x;
    const int l   = tid & 31;
    const int w   = tid >> 5;
    const int gw  = blockIdx.x * 8 + w;     // 0..4095
    const int tm  = gw & 255;               // M tile (16 rows each)
    const int tn  = gw >> 8;                // N super-tile (64 cols each)
    const int lc  = l & 15;
    const int rbase = (l < 16) ? 0 : 8;
    const int colbase = tn * 64;

    v8f acc[4];
#pragma unroll
    for (int i = 0; i < 4; ++i)
#pragma unroll
        for (int j = 0; j < 8; ++j) acc[i][j] = 0.0f;

    // swizzled bases: row-tile stride = 32 tiles * 512 elems
    const __bf16* Abase = A + (size_t)tm * (32 * 512) + l * 16;
    const __bf16* Wbase[4];
#pragma unroll
    for (int sub = 0; sub < 4; ++sub)
        Wbase[sub] = W + (size_t)(tn * 4 + sub) * (32 * 512) + l * 16;

    // prologue: fill ping-pong buffers with K-steps 0 and 1
    v16bf afA = *(const v16bf*)(Abase);
    v16bf bfA[4];
#pragma unroll
    for (int sub = 0; sub < 4; ++sub) bfA[sub] = *(const v16bf*)(Wbase[sub]);
    v16bf afB = *(const v16bf*)(Abase + 512);
    v16bf bfB[4];
#pragma unroll
    for (int sub = 0; sub < 4; ++sub) bfB[sub] = *(const v16bf*)(Wbase[sub] + 512);

    // steady state: loads write straight into ping-pong buffers (no copies)
    for (int ks = 0; ks < 30; ks += 2) {
#pragma unroll
        for (int sub = 0; sub < 4; ++sub)
            acc[sub] = wmma_bf16(afA, bfA[sub], acc[sub]);
        afA = *(const v16bf*)(Abase + (size_t)(ks + 2) * 512);
#pragma unroll
        for (int sub = 0; sub < 4; ++sub)
            bfA[sub] = *(const v16bf*)(Wbase[sub] + (size_t)(ks + 2) * 512);
#pragma unroll
        for (int sub = 0; sub < 4; ++sub)
            acc[sub] = wmma_bf16(afB, bfB[sub], acc[sub]);
        afB = *(const v16bf*)(Abase + (size_t)(ks + 3) * 512);
#pragma unroll
        for (int sub = 0; sub < 4; ++sub)
            bfB[sub] = *(const v16bf*)(Wbase[sub] + (size_t)(ks + 3) * 512);
    }
    // tail: K-steps 30 and 31
#pragma unroll
    for (int sub = 0; sub < 4; ++sub)
        acc[sub] = wmma_bf16(afA, bfA[sub], acc[sub]);
#pragma unroll
    for (int sub = 0; sub < 4; ++sub)
        acc[sub] = wmma_bf16(afB, bfB[sub], acc[sub]);

    // Epilogue: bias + GELU (+RoPE) + remapped swizzled store.
#pragma unroll
    for (int sub = 0; sub < 4; ++sub) {
        const int col = colbase + sub * 16 + lc;
        const float bv = bias[col];
        const int j = (col & 63) >> 1;     // theta depends only on column
        const float theta = __expf(-(float)(2 * j) * (9.210340371976184f / 64.0f));
#pragma unroll
        for (int r = 0; r < 8; ++r) {
            const int mrow = tm * 16 + r + rbase;   // C layout: rows r / r+8
            float x = acc[sub][r] + bv;
            x = gelu_exact(x);
            const int s = mrow & 2047;
            const int bb = mrow >> 11;
            if (do_rope) {
                float sn, cs;
                __sincosf((float)s * theta, &sn, &cs);
                const float xp = __shfl_xor(x, 1, 32);   // even/odd dim partner
                x = (l & 1) ? (x * cs + xp * sn) : (x * cs - xp * sn);
            }
            if (mode == 0) {        // Qb/Kb: per-head [t][64] swizzled
                const int n = ((s & 1) << 4) + (col >> 6);
                const int t = (bb << 10) + (s >> 1);
                out_bf[(size_t)n * 131072 + swz_idx(t, col & 63, 64)] = (__bf16)x;
            } else if (mode == 1) { // Vt: per-head [64][t] swizzled
                const int n = ((s & 1) << 4) + (col >> 6);
                const int t = (bb << 10) + (s >> 1);
                out_bf[(size_t)n * 131072 + swz_idx(col & 63, t, 2048)] = (__bf16)x;
            } else {                // final fp32, identity onto (B,S,D)
                out_f[(size_t)mrow * 1024 + col] = x;
            }
        }
    }
}

// ---------------------------------------------------------------------------
// Flash attention: 1 wave per (group n, 16-query tile), stream 32 keys/iter.
// scores = Q @ K^T * (1/32); online softmax fp32; O += P @ V via WMMA.
// All Q/K/V fragments are single 32B loads from swizzled buffers; P is
// transposed C->A layout through a swizzled per-wave LDS tile (1KB).
// ---------------------------------------------------------------------------
__global__ __launch_bounds__(128) void attn_kernel(
    const __bf16* __restrict__ Q, const __bf16* __restrict__ K,
    const __bf16* __restrict__ Vt, __bf16* __restrict__ Oout) {
    __shared__ __bf16 lds[4][16 * 32];
    const int tid = threadIdx.x;
    const int l   = tid & 31;
    const int w   = tid >> 5;
    const int gw  = blockIdx.x * 4 + w;       // 0..4095
    const int n   = gw >> 7;                  // attention group 0..31
    const int q0  = (gw & 127) * 16;          // query tile start
    const int lc  = l & 15;
    const int rbase = (l < 16) ? 0 : 8;
    const float SCALE = 0.03125f;             // sqrt(1/D) = 1/32

    const __bf16* Qh = Q  + (size_t)n * 131072 + l * 16;  // [t][64] swizzled
    const __bf16* Kh = K  + (size_t)n * 131072 + l * 16;  // [t][64] swizzled
    const __bf16* Vh = Vt + (size_t)n * 131072 + l * 16;  // [64][t] swizzled

    // Q fragments: tile = (q0>>4)*2 + c
    v16bf aQ[2];
#pragma unroll
    for (int c = 0; c < 2; ++c)
        aQ[c] = *(const v16bf*)(Qh + (size_t)((q0 >> 4) * 2 + c) * 512);

    float mrun[8], lrun[8];
    v8f og[4];
#pragma unroll
    for (int r = 0; r < 8; ++r) { mrun[r] = -1e30f; lrun[r] = 0.0f; }
#pragma unroll
    for (int g = 0; g < 4; ++g)
#pragma unroll
        for (int r = 0; r < 8; ++r) og[g][r] = 0.0f;

    for (int k0 = 0; k0 < 2048; k0 += 32) {
        // ---- issue ALL global loads for this k-tile up front ----
        const int kt = k0 >> 4;
        v16bf bK0[2], bK1[2];
#pragma unroll
        for (int c = 0; c < 2; ++c) {
            bK0[c] = *(const v16bf*)(Kh + (size_t)(kt * 2 + c) * 512);
            bK1[c] = *(const v16bf*)(Kh + (size_t)((kt + 1) * 2 + c) * 512);
        }
        v16bf bV[4];
#pragma unroll
        for (int g = 0; g < 4; ++g)   // Vt tile = g*64 + (k0>>5)
            bV[g] = *(const v16bf*)(Vh + (size_t)(g * 64 + (k0 >> 5)) * 512);

        // ---- scores: two 16-key column groups, K=64 in 2 steps ----
        v8f sa0, sa1;
#pragma unroll
        for (int r = 0; r < 8; ++r) { sa0[r] = 0.0f; sa1[r] = 0.0f; }
#pragma unroll
        for (int c = 0; c < 2; ++c) {
            sa0 = wmma_bf16(aQ[c], bK0[c], sa0);
            sa1 = wmma_bf16(aQ[c], bK1[c], sa1);
        }

        // ---- online softmax update (row stats via 16-lane xor reductions)
        float p0[8], p1[8], alpha[8];
#pragma unroll
        for (int r = 0; r < 8; ++r) {
            const float s0 = sa0[r] * SCALE;
            const float s1 = sa1[r] * SCALE;
            float mx = fmaxf(s0, s1);
            mx = fmaxf(mx, __shfl_xor(mx, 1, 32));
            mx = fmaxf(mx, __shfl_xor(mx, 2, 32));
            mx = fmaxf(mx, __shfl_xor(mx, 4, 32));
            mx = fmaxf(mx, __shfl_xor(mx, 8, 32));
            const float mnew = fmaxf(mrun[r], mx);
            const float a = __expf(mrun[r] - mnew);
            mrun[r] = mnew;
            const float e0 = __expf(s0 - mnew);
            const float e1 = __expf(s1 - mnew);
            float rs = e0 + e1;
            rs += __shfl_xor(rs, 1, 32);
            rs += __shfl_xor(rs, 2, 32);
            rs += __shfl_xor(rs, 4, 32);
            rs += __shfl_xor(rs, 8, 32);
            lrun[r] = lrun[r] * a + rs;
            p0[r] = e0; p1[r] = e1; alpha[r] = a;
        }
#pragma unroll
        for (int g = 0; g < 4; ++g)
#pragma unroll
            for (int r = 0; r < 8; ++r) og[g][r] *= alpha[r];

        // ---- transpose P (C -> A layout) through swizzled LDS tile ----
        // writer lane holds cols lc (p0) and 16+lc (p1) of rows rbase+r;
        // swizzled: lane_r = row + 16*((lc>>3)&1), elems (lc&7) and (lc&7)+8
        {
            const int lhi = ((lc >> 3) & 1) << 4;
            const int eP  = lc & 7;
#pragma unroll
            for (int r = 0; r < 8; ++r) {
                const int lrP = (rbase + r) + lhi;
                lds[w][lrP * 16 + eP]     = (__bf16)p0[r];
                lds[w][lrP * 16 + eP + 8] = (__bf16)p1[r];
            }
        }
        asm volatile("s_wait_dscnt 0" ::: "memory");
        v16bf aP = *(const v16bf*)(&lds[w][l * 16]);   // single 32B LDS read
        asm volatile("" ::: "memory");

        // ---- O += P @ V (V fragments already resident) ----
#pragma unroll
        for (int g = 0; g < 4; ++g)
            og[g] = wmma_bf16(aP, bV[g], og[g]);
    }

    // finalize: divide by row sums, scatter to swizzled AttnS (A of final GEMM)
#pragma unroll
    for (int r = 0; r < 8; ++r) {
        const float inv = 1.0f / lrun[r];
        const int t = q0 + rbase + r;
        const int rout = 2 * t + (n >> 4);
#pragma unroll
        for (int g = 0; g < 4; ++g) {
            const int colo = ((n & 15) << 6) + g * 16 + lc;
            Oout[swz_idx(rout, colo, 1024)] = (__bf16)(og[g][r] * inv);
        }
    }
}

// ---------------------------------------------------------------------------
extern "C" void kernel_launch(void* const* d_in, const int* in_sizes, int n_in,
                              void* d_out, int out_size, void* d_ws, size_t ws_size,
                              hipStream_t stream) {
    const float* q  = (const float*)d_in[0];
    const float* k  = (const float*)d_in[1];
    const float* v  = (const float*)d_in[2];
    const float* Wq = (const float*)d_in[3];
    const float* bq = (const float*)d_in[4];
    const float* Wk = (const float*)d_in[5];
    const float* bk = (const float*)d_in[6];
    const float* Wv = (const float*)d_in[7];
    const float* bv = (const float*)d_in[8];
    const float* Wo = (const float*)d_in[9];
    const float* bo = (const float*)d_in[10];
    float* out = (float*)d_out;

    const size_t NW = 1024u * 1024u;   // weight elements
    const size_t NX = 4096u * 1024u;   // activation elements

    char* ws = (char*)d_ws;
    size_t off = 0;
    auto carve = [&](size_t bytes) -> char* {
        char* p = ws + off;
        off += (bytes + 255) & ~(size_t)255;
        return p;
    };
    __bf16* Wq_b = (__bf16*)carve(NW * 2);
    __bf16* Wk_b = (__bf16*)carve(NW * 2);
    __bf16* Wv_b = (__bf16*)carve(NW * 2);
    __bf16* Wo_b = (__bf16*)carve(NW * 2);
    __bf16* q_b  = (__bf16*)carve(NX * 2);
    __bf16* k_b  = (__bf16*)carve(NX * 2);
    __bf16* v_b  = (__bf16*)carve(NX * 2);
    __bf16* Qb   = (__bf16*)carve(NX * 2);
    __bf16* Kb   = (__bf16*)carve(NX * 2);
    __bf16* Vt   = (__bf16*)carve(NX * 2);
    __bf16* AttnS = q_b;   // alias: q_b is dead once proj-q has run

    // 1) fp32 -> bf16 swizzled staging (8-col chunk per thread)
    {
        const int tpb = 256;
        int ncX = (int)(NX / 8), ncW = (int)(NW / 8);
        cvt_swz<<<(ncX + tpb - 1) / tpb, tpb, 0, stream>>>(q, q_b, ncX);
        cvt_swz<<<(ncX + tpb - 1) / tpb, tpb, 0, stream>>>(k, k_b, ncX);
        cvt_swz<<<(ncX + tpb - 1) / tpb, tpb, 0, stream>>>(v, v_b, ncX);
        cvt_swz<<<(ncW + tpb - 1) / tpb, tpb, 0, stream>>>(Wq, Wq_b, ncW);
        cvt_swz<<<(ncW + tpb - 1) / tpb, tpb, 0, stream>>>(Wk, Wk_b, ncW);
        cvt_swz<<<(ncW + tpb - 1) / tpb, tpb, 0, stream>>>(Wv, Wv_b, ncW);
        cvt_swz<<<(ncW + tpb - 1) / tpb, tpb, 0, stream>>>(Wo, Wo_b, ncW);
    }

    // 2) projections + GELU (+RoPE), remapped swizzled stores
    proj_gemm<<<512, 256, 0, stream>>>(q_b, Wq_b, bq, Qb, nullptr, 0, 1);
    proj_gemm<<<512, 256, 0, stream>>>(k_b, Wk_b, bk, Kb, nullptr, 0, 1);
    proj_gemm<<<512, 256, 0, stream>>>(v_b, Wv_b, bv, Vt, nullptr, 1, 0);

    // 3) flash attention (32 groups x 128 query tiles = 4096 waves)
    attn_kernel<<<1024, 128, 0, stream>>>(Qb, Kb, Vt, AttnS);

    // 4) output projection + GELU -> fp32 (B,S,D)
    proj_gemm<<<512, 256, 0, stream>>>(AttnS, Wo_b, bo, nullptr, out, 2, 0);

    (void)in_sizes; (void)n_in; (void)out_size; (void)ws_size;
}
